// BasicAttention_52123723104341
// MI455X (gfx1250) — compile-verified, matrix operand order checked by reading, not run on hardware
//
#include <hip/hip_runtime.h>
#include <hip/hip_bf16.h>

// ---------------------------------------------------------------------------
// BasicAttention for MI455X (gfx1250): bf16 WMMA + async global->LDS DMA.
//   S=4096, HID=2048, H=16, HKV=8, D=128
// Pipeline:
//   1) hidden fp32 -> bf16
//   2) W{q,k,v,o} fp32 [K][N] -> bf16 [N][K] (transposed for B-fragments)
//   3) bf16 GEMM (v_wmma_f32_16x16x32_bf16, async-to-LDS double buffering)
//   4) RMSNorm+RoPE fused (fp32 VALU) -> qb/kb bf16 head-major; V -> vt [h][d][s]
//   5) flash attention (WMMA QK^T + WMMA PV, online softmax, async LDS fill)
//   6) bf16 GEMM: out = ob x Wo^T  (fp32 result)
// ---------------------------------------------------------------------------

#define S_LEN 4096
#define HID   2048
#define NH    16
#define NKV   8
#define HD    128

typedef __bf16 bf16x16 __attribute__((ext_vector_type(16)));
typedef __bf16 bf16x8  __attribute__((ext_vector_type(8)));
typedef __bf16 bf16x4  __attribute__((ext_vector_type(4)));
typedef float  f32x8   __attribute__((ext_vector_type(8)));

union FragU { bf16x16 v; bf16x8 h[2]; };

// Low 32 bits of the shared-aperture generic address == LDS byte address.
__device__ __forceinline__ unsigned lds_addr(const void* p) {
    return (unsigned)(uintptr_t)p;
}

// CDNA5 async DMA: 16B global -> LDS, tracked by ASYNCcnt (no VGPR staging).
__device__ __forceinline__ void async_load_b128(unsigned lds, const void* gaddr) {
    asm volatile("global_load_async_to_lds_b128 %0, %1, off"
                 :: "v"(lds), "v"(gaddr) : "memory");
}
__device__ __forceinline__ void wait_async0() {
    asm volatile("s_wait_asynccnt 0" ::: "memory");
}

// Load one 16-bit WMMA A/B fragment (16x32 / 32x16) for this lane.
// Per ISA 7.12.2: elements 0..7  -> K = kc*32 + half*8 + [0..8),
//                 elements 8..15 -> K = kc*32 + 16 + half*8 + [0..8).
__device__ __forceinline__ bf16x16 ld_frag(const __bf16* rowp, int kc, int lh) {
    FragU u;
    const __bf16* p = rowp + kc * 32 + lh * 8;
    u.h[0] = *(const bf16x8*)(p);
    u.h[1] = *(const bf16x8*)(p + 16);
    return u.v;
}

// ---------------------------------------------------------------------------
// 1) fp32 -> bf16 elementwise (vector x4)
// ---------------------------------------------------------------------------
__global__ __launch_bounds__(256) void k_convert_bf16(const float* __restrict__ in,
                                                      __bf16* __restrict__ out, int n4) {
    int i = blockIdx.x * 256 + threadIdx.x;
    if (i >= n4) return;
    float4 v = *(const float4*)(in + (size_t)i * 4);
    bf16x4 o;
    o[0] = (__bf16)v.x; o[1] = (__bf16)v.y; o[2] = (__bf16)v.z; o[3] = (__bf16)v.w;
    *(bf16x4*)(out + (size_t)i * 4) = o;
}

// ---------------------------------------------------------------------------
// 2) transpose+convert: out[n][k] = (bf16) in[k][n].  K,N multiples of 32.
// ---------------------------------------------------------------------------
__global__ __launch_bounds__(256) void k_transpose_bf16(const float* __restrict__ in,
                                                        __bf16* __restrict__ out,
                                                        int K, int N) {
    __shared__ float t[32][33];
    int k0 = blockIdx.y * 32, n0 = blockIdx.x * 32;
    int tx = threadIdx.x, ty = threadIdx.y;
#pragma unroll
    for (int y = 0; y < 4; ++y)
        t[ty + y * 8][tx] = in[(size_t)(k0 + ty + y * 8) * N + n0 + tx];
    __syncthreads();
#pragma unroll
    for (int y = 0; y < 4; ++y)
        out[(size_t)(n0 + ty + y * 8) * K + k0 + tx] = (__bf16)t[tx][ty + y * 8];
}

// ---------------------------------------------------------------------------
// 3) bf16 GEMM: C[M][N] (fp32) = A[M][K] (bf16) * Bt[N][K] (bf16)
//    BM=128, BN=64, BK=64; 256 threads = 8 waves; wave w -> 16 rows, 64 cols.
//    Double-buffered LDS, filled by async global->LDS DMA.
// ---------------------------------------------------------------------------
__global__ __launch_bounds__(256) void k_gemm_bf16(const __bf16* __restrict__ A,
                                                   const __bf16* __restrict__ Bt,
                                                   float* __restrict__ C,
                                                   int M, int N, int K) {
    __shared__ __bf16 As[2][128][64];   // 2 x 16 KB
    __shared__ __bf16 Bs[2][64][64];    // 2 x  8 KB
    const int m0 = blockIdx.y * 128, n0 = blockIdx.x * 64;
    const int tid = threadIdx.x, w = tid >> 5, lane = tid & 31;
    const int lh = lane >> 4, l16 = lane & 15;

    // per-thread tile-fill assignment (16B chunks)
    const int ra = tid >> 3, ca = (tid & 7) * 8;        // A: 4 chunks/thread
    const int rb = tid >> 3, cb = (tid & 7) * 8;        // B: 2 chunks/thread

    auto issue_tile = [&](int buf, int k0) {
#pragma unroll
        for (int i = 0; i < 4; ++i) {
            int r = ra + i * 32;
            async_load_b128(lds_addr(&As[buf][r][ca]),
                            &A[(size_t)(m0 + r) * K + k0 + ca]);
        }
#pragma unroll
        for (int i = 0; i < 2; ++i) {
            int r = rb + i * 32;
            if (r < 64)
                async_load_b128(lds_addr(&Bs[buf][r][cb]),
                                &Bt[(size_t)(n0 + r) * K + k0 + cb]);
        }
    };

    f32x8 acc[4];
#pragma unroll
    for (int j = 0; j < 4; ++j)
#pragma unroll
        for (int e = 0; e < 8; ++e) acc[j][e] = 0.f;

    int cur = 0;
    issue_tile(0, 0);
    for (int k0 = 0; k0 < K; k0 += 64) {
        wait_async0();          // my DMA into As/Bs[cur] has landed
        __syncthreads();        // everyone's has landed; prev buffer free
        if (k0 + 64 < K) issue_tile(cur ^ 1, k0 + 64);

        // batch all fragment loads, then the WMMA chain
        bf16x16 afr[2], bfr[8];
#pragma unroll
        for (int kc = 0; kc < 2; ++kc)
            afr[kc] = ld_frag(&As[cur][w * 16 + l16][0], kc, lh);
#pragma unroll
        for (int j = 0; j < 4; ++j)
#pragma unroll
            for (int kc = 0; kc < 2; ++kc)
                bfr[j * 2 + kc] = ld_frag(&Bs[cur][j * 16 + l16][0], kc, lh);
#pragma unroll
        for (int kc = 0; kc < 2; ++kc)
#pragma unroll
            for (int j = 0; j < 4; ++j)
                acc[j] = __builtin_amdgcn_wmma_f32_16x16x32_bf16(
                    false, afr[kc], false, bfr[j * 2 + kc], (short)0, acc[j],
                    false, false);
        __syncthreads();        // done reading buf[cur] before it is refilled
        cur ^= 1;
    }
#pragma unroll
    for (int j = 0; j < 4; ++j)
#pragma unroll
        for (int r = 0; r < 8; ++r) {
            int row = m0 + w * 16 + lh * 8 + r;
            int col = n0 + j * 16 + l16;
            C[(size_t)row * N + col] = acc[j][r];
        }
}

// ---------------------------------------------------------------------------
// 4a) fused RMSNorm + RoPE.  One wave handles one (s, head) row of 128.
//     in:  x fp32 [S][nH*128];  out: bf16 [nH][S][128] (head-major)
// ---------------------------------------------------------------------------
__global__ __launch_bounds__(256) void k_rmsnorm_rope(const float* __restrict__ x,
                                                      const float* __restrict__ nw,
                                                      const int* __restrict__ pos,
                                                      __bf16* __restrict__ out,
                                                      int nH) {
    const int lane = threadIdx.x & 31;
    const int wv = threadIdx.x >> 5;
    const int rowid = blockIdx.x * 8 + wv;      // 0 .. S*nH-1
    const int s = rowid / nH, h = rowid % nH;
    const float* xr = x + (size_t)(s * nH + h) * HD;
    const int d0 = lane * 4;

    float4 v = *(const float4*)(xr + d0);
    float a[4] = {v.x, v.y, v.z, v.w};
    float ss = a[0]*a[0] + a[1]*a[1] + a[2]*a[2] + a[3]*a[3];
#pragma unroll
    for (int m = 1; m < 32; m <<= 1) ss += __shfl_xor(ss, m, 32);
    const float inv = rsqrtf(ss * (1.0f / 128.0f) + 1e-6f);
#pragma unroll
    for (int e = 0; e < 4; ++e) a[e] = a[e] * inv * nw[d0 + e];

    const float p = (float)pos[s];
    const int j0 = (lane & 15) * 4;             // rotary pair index
    bf16x4 o;
#pragma unroll
    for (int e = 0; e < 4; ++e) {
        float oth = __shfl_xor(a[e], 16, 32);   // partner half of the rotation
        float ang = p * expf(-(float)(j0 + e) * 0.14391156831212787f); // ln(1e4)/64
        float c, sn;
        __sincosf(ang, &sn, &c);
        o[e] = (__bf16)((lane < 16) ? a[e] * c - oth * sn
                                    : a[e] * c + oth * sn);
    }
    *(bf16x4*)(out + ((size_t)h * S_LEN + s) * HD + d0) = o;
}

// ---------------------------------------------------------------------------
// 4b) V transpose: vt[h][d][s] = (bf16) vf[s][h*128+d]
// ---------------------------------------------------------------------------
__global__ __launch_bounds__(256) void k_v_transpose(const float* __restrict__ vf,
                                                     __bf16* __restrict__ vt) {
    __shared__ float t[32][33];
    const int h = blockIdx.z, s0 = blockIdx.x * 32, dd0 = blockIdx.y * 32;
    const int tx = threadIdx.x, ty = threadIdx.y;
#pragma unroll
    for (int y = 0; y < 4; ++y)
        t[ty + y * 8][tx] = vf[(size_t)(s0 + ty + y * 8) * (NKV * HD) + h * HD + dd0 + tx];
    __syncthreads();
#pragma unroll
    for (int y = 0; y < 4; ++y)
        vt[((size_t)h * HD + dd0 + ty + y * 8) * S_LEN + s0 + tx] = (__bf16)t[tx][ty + y * 8];
}

// ---------------------------------------------------------------------------
// 5) Flash attention. Block = 128 threads (4 waves); block covers 64 q rows
//    of one head, full D=128. KV tiles of 64 filled by async DMA.
// ---------------------------------------------------------------------------
__global__ __launch_bounds__(128) void k_flash_attn(const __bf16* __restrict__ qb,
                                                    const __bf16* __restrict__ kb,
                                                    const __bf16* __restrict__ vt,
                                                    __bf16* __restrict__ ob) {
    __shared__ __bf16 Ks[64][128];   // [kv][d]  -> B-frags for QK^T (N=kv, K=d)
    __shared__ __bf16 Vs[128][64];   // [d][kv]  -> B-frags for PV   (N=d,  K=kv)
    __shared__ __bf16 Ps[4][16][64]; // per-wave P tile (A operand of PV)

    const int qt = blockIdx.x;       // 64-row q tile
    const int h  = blockIdx.y;
    const int hkv = h >> 1;          // GQA: 2 q heads per kv head
    const int tid = threadIdx.x, w = tid >> 5, lane = tid & 31;
    const int lh = lane >> 4, l16 = lane & 15;

    // Q fragments: rows qt*64 + w*16 + (lane%16), D split into 4 K-chunks of 32
    const __bf16* qrowp = qb + ((size_t)h * S_LEN + qt * 64 + w * 16 + l16) * HD;
    bf16x16 aq[4];
#pragma unroll
    for (int kc = 0; kc < 4; ++kc) aq[kc] = ld_frag(qrowp, kc, lh);

    f32x8 acc[8];
#pragma unroll
    for (int j = 0; j < 8; ++j)
#pragma unroll
        for (int e = 0; e < 8; ++e) acc[j][e] = 0.f;
    float m_r[8], l_r[8], alpha[8];
#pragma unroll
    for (int r = 0; r < 8; ++r) { m_r[r] = -3.0e38f; l_r[r] = 0.f; }

    const float SC = 0.08838834764831845f * 1.4426950408889634f; // 1/sqrt(128)*log2(e)
    const int qhi = qt * 64;
    const __bf16* kbase = kb + (size_t)hkv * S_LEN * HD;
    const __bf16* vbase = vt + (size_t)hkv * HD * S_LEN;

    for (int kj = 0; kj <= qhi; kj += 64) {
        __syncthreads();             // everyone done reading the previous tile
        // Ks: 64x128 bf16 = 1024 16B-chunks; 8 per thread (async DMA)
#pragma unroll
        for (int i = 0; i < 8; ++i) {
            int idx = tid + i * 128;
            int r = idx >> 4, c = (idx & 15) * 8;
            async_load_b128(lds_addr(&Ks[r][c]),
                            kbase + (size_t)(kj + r) * HD + c);
        }
        // Vs: 128x64 bf16 = 1024 16B-chunks; 8 per thread
#pragma unroll
        for (int i = 0; i < 8; ++i) {
            int idx = tid + i * 128;
            int r = idx >> 3, c = (idx & 7) * 8;
            async_load_b128(lds_addr(&Vs[r][c]),
                            vbase + (size_t)r * S_LEN + kj + c);
        }
        wait_async0();
        __syncthreads();             // tile fully in LDS

        // ---- scores = Q x K^T ------------------------------------------------
        f32x8 sc[4];
#pragma unroll
        for (int j = 0; j < 4; ++j)
#pragma unroll
            for (int e = 0; e < 8; ++e) sc[j][e] = 0.f;
#pragma unroll
        for (int kc = 0; kc < 4; ++kc) {
            bf16x16 bk[4];
#pragma unroll
            for (int nj = 0; nj < 4; ++nj)
                bk[nj] = ld_frag(&Ks[nj * 16 + l16][0], kc, lh);
#pragma unroll
            for (int nj = 0; nj < 4; ++nj)
                sc[nj] = __builtin_amdgcn_wmma_f32_16x16x32_bf16(
                    false, aq[kc], false, bk[nj], (short)0, sc[nj], false, false);
        }

        // ---- causal mask + online softmax -----------------------------------
        float rmax[8];
#pragma unroll
        for (int r = 0; r < 8; ++r) rmax[r] = -3.0e38f;
#pragma unroll
        for (int nj = 0; nj < 4; ++nj)
#pragma unroll
            for (int r = 0; r < 8; ++r) {
                float v = sc[nj][r] * SC;
                int kidx = kj + nj * 16 + l16;
                int qidx = qt * 64 + w * 16 + lh * 8 + r;
                v = (kidx > qidx) ? -3.0e38f : v;
                sc[nj][r] = v;
                rmax[r] = fmaxf(rmax[r], v);
            }
#pragma unroll
        for (int r = 0; r < 8; ++r) {
            float rm = rmax[r];
            rm = fmaxf(rm, __shfl_xor(rm, 1, 32));
            rm = fmaxf(rm, __shfl_xor(rm, 2, 32));
            rm = fmaxf(rm, __shfl_xor(rm, 4, 32));
            rm = fmaxf(rm, __shfl_xor(rm, 8, 32));
            float mn = fmaxf(m_r[r], rm);
            alpha[r] = exp2f(m_r[r] - mn);
            m_r[r] = mn;
            l_r[r] *= alpha[r];
        }
        float rsum[8];
#pragma unroll
        for (int r = 0; r < 8; ++r) rsum[r] = 0.f;
#pragma unroll
        for (int nj = 0; nj < 4; ++nj)
#pragma unroll
            for (int r = 0; r < 8; ++r) {
                float pv = exp2f(sc[nj][r] - m_r[r]);
                rsum[r] += pv;
                Ps[w][lh * 8 + r][nj * 16 + l16] = (__bf16)pv;
            }
#pragma unroll
        for (int r = 0; r < 8; ++r) {
            float sm = rsum[r];
            sm += __shfl_xor(sm, 1, 32);
            sm += __shfl_xor(sm, 2, 32);
            sm += __shfl_xor(sm, 4, 32);
            sm += __shfl_xor(sm, 8, 32);
            l_r[r] += sm;
        }
#pragma unroll
        for (int j = 0; j < 8; ++j)
#pragma unroll
            for (int r = 0; r < 8; ++r) acc[j][r] *= alpha[r];

        // wave-local LDS RAW on Ps: explicit split-counter wait (no barrier)
        asm volatile("s_wait_dscnt 0" ::: "memory");

        // ---- acc += P x V ----------------------------------------------------
#pragma unroll
        for (int kc = 0; kc < 2; ++kc) {
            bf16x16 ap = ld_frag(&Ps[w][l16][0], kc, lh);
#pragma unroll
            for (int g = 0; g < 2; ++g) {
                bf16x16 bv[4];
#pragma unroll
                for (int t = 0; t < 4; ++t)
                    bv[t] = ld_frag(&Vs[(g * 4 + t) * 16 + l16][0], kc, lh);
#pragma unroll
                for (int t = 0; t < 4; ++t)
                    acc[g * 4 + t] = __builtin_amdgcn_wmma_f32_16x16x32_bf16(
                        false, ap, false, bv[t], (short)0, acc[g * 4 + t],
                        false, false);
            }
        }
    }

    // ---- normalize + store bf16 for the output projection --------------------
#pragma unroll
    for (int r = 0; r < 8; ++r) {
        const int row = qt * 64 + w * 16 + lh * 8 + r;
        const float invl = 1.0f / l_r[r];
#pragma unroll
        for (int dj = 0; dj < 8; ++dj)
            ob[(size_t)row * (NH * HD) + h * HD + dj * 16 + l16] =
                (__bf16)(acc[dj][r] * invl);
    }
}

// ---------------------------------------------------------------------------
// Host launcher
// ---------------------------------------------------------------------------
extern "C" void kernel_launch(void* const* d_in, const int* in_sizes, int n_in,
                              void* d_out, int out_size, void* d_ws, size_t ws_size,
                              hipStream_t stream) {
    (void)in_sizes; (void)n_in; (void)out_size; (void)ws_size;
    const float* hidden = (const float*)d_in[0];
    const float* Wq = (const float*)d_in[1];
    const float* Wk = (const float*)d_in[2];
    const float* Wv = (const float*)d_in[3];
    const float* Wo = (const float*)d_in[4];
    const float* qnw = (const float*)d_in[5];
    const float* knw = (const float*)d_in[6];
    const int*   pos = (const int*)d_in[7];
    float* out = (float*)d_out;

    char* wp = (char*)d_ws;
    auto alloc = [&](size_t bytes) -> void* {
        void* p = (void*)wp;
        wp += (bytes + 255) & ~(size_t)255;
        return p;
    };
    __bf16* hb  = (__bf16*)alloc((size_t)S_LEN * HID * 2);
    __bf16* Wqt = (__bf16*)alloc((size_t)HID * (NH * HD) * 2);
    __bf16* Wkt = (__bf16*)alloc((size_t)HID * (NKV * HD) * 2);
    __bf16* Wvt = (__bf16*)alloc((size_t)HID * (NKV * HD) * 2);
    __bf16* Wot = (__bf16*)alloc((size_t)HID * (NH * HD) * 2);
    float*  qf  = (float*)alloc((size_t)S_LEN * NH * HD * 4);
    float*  kf  = (float*)alloc((size_t)S_LEN * NKV * HD * 4);
    float*  vf  = (float*)alloc((size_t)S_LEN * NKV * HD * 4);
    __bf16* qb  = (__bf16*)alloc((size_t)NH * S_LEN * HD * 2);
    __bf16* kb  = (__bf16*)alloc((size_t)NKV * S_LEN * HD * 2);
    __bf16* vt  = (__bf16*)alloc((size_t)NKV * HD * S_LEN * 2);
    __bf16* ob  = (__bf16*)alloc((size_t)S_LEN * NH * HD * 2);

    // 1) hidden -> bf16
    k_convert_bf16<<<(S_LEN * HID / 4) / 256, 256, 0, stream>>>(hidden, hb, S_LEN * HID / 4);

    // 2) weights -> transposed bf16  (in is [K][N] row-major, out [N][K])
    k_transpose_bf16<<<dim3((NH * HD) / 32, HID / 32), dim3(32, 8), 0, stream>>>(Wq, Wqt, HID, NH * HD);
    k_transpose_bf16<<<dim3((NKV * HD) / 32, HID / 32), dim3(32, 8), 0, stream>>>(Wk, Wkt, HID, NKV * HD);
    k_transpose_bf16<<<dim3((NKV * HD) / 32, HID / 32), dim3(32, 8), 0, stream>>>(Wv, Wvt, HID, NKV * HD);
    k_transpose_bf16<<<dim3(HID / 32, (NH * HD) / 32), dim3(32, 8), 0, stream>>>(Wo, Wot, NH * HD, HID);

    // 3) QKV projections
    k_gemm_bf16<<<dim3((NH * HD) / 64, S_LEN / 128), 256, 0, stream>>>(hb, Wqt, qf, S_LEN, NH * HD, HID);
    k_gemm_bf16<<<dim3((NKV * HD) / 64, S_LEN / 128), 256, 0, stream>>>(hb, Wkt, kf, S_LEN, NKV * HD, HID);
    k_gemm_bf16<<<dim3((NKV * HD) / 64, S_LEN / 128), 256, 0, stream>>>(hb, Wvt, vf, S_LEN, NKV * HD, HID);

    // 4) RMSNorm + RoPE (q, k) and V transpose
    k_rmsnorm_rope<<<(S_LEN * NH) / 8, 256, 0, stream>>>(qf, qnw, pos, qb, NH);
    k_rmsnorm_rope<<<(S_LEN * NKV) / 8, 256, 0, stream>>>(kf, knw, pos, kb, NKV);
    k_v_transpose<<<dim3(S_LEN / 32, HD / 32, NKV), dim3(32, 8), 0, stream>>>(vf, vt);

    // 5) flash attention
    k_flash_attn<<<dim3(S_LEN / 64, NH), 128, 0, stream>>>(qb, kb, vt, ob);

    // 6) output projection
    k_gemm_bf16<<<dim3(HID / 64, S_LEN / 128), 256, 0, stream>>>(ob, Wot, out, S_LEN, HID, NH * HD);
}
